// DeformConv2D_48773648613600
// MI455X (gfx1250) — matrix operand order repaired and checked by reading, not compiled, not run on hardware
//
#include <hip/hip_runtime.h>
#include <stdint.h>

// ---------------- CDNA5 WMMA / vector types ----------------
typedef __attribute__((ext_vector_type(16))) __bf16        v16bf;
typedef __attribute__((ext_vector_type(8)))  float         v8f;
typedef __attribute__((ext_vector_type(4)))  unsigned int  u32x4;

// gfx1250 async global->LDS copy (probe via __has_builtin; fallback = plain copy)
#if defined(__gfx1250__) && __has_builtin(__builtin_amdgcn_global_load_async_to_lds_b128) && \
    __has_builtin(__builtin_amdgcn_s_wait_asynccnt)
#define HAVE_ASYNC_LDS 1
typedef __attribute__((__vector_size__(16))) int           i32x4v;    // builtin's pointee type
typedef __attribute__((address_space(3)))    i32x4v        as3_i32x4v; // LDS-space variant
#else
#define HAVE_ASYNC_LDS 0
#endif

// ---------------- problem constants ----------------
constexpr int kBat  = 4;
constexpr int kInC  = 128;
constexpr int kOutC = 128;
constexpr int kH    = 96;
constexpr int kW    = 96;
constexpr int kHW   = kH * kW;          // 9216
constexpr int kTaps = 9;                // 3x3
constexpr int kK    = kTaps * kInC;     // 1152 (GEMM K, k = n*128 + c)
constexpr int kKd   = kK / 2;           // 576 dwords per bf16 weight row
constexpr int kChunks = kK / 32;        // 36 K-chunks of 32
constexpr int kNT   = 32;               // main-conv pixels per block
constexpr int kNTo  = 64;               // offset-conv pixels per block
constexpr int kStr  = 20;               // LDS row stride (dwords): 16B aligned, conflict-free

// meta: 4 precomputed plane offsets + 4 pad-masked bilinear weights (32 B)
struct Meta { int a00, a11, a01, a10; float w0, w1, w2, w3; };

__device__ __forceinline__ unsigned int f2bf(float f) {
  union { float f; unsigned int u; } cv; cv.f = f;
  unsigned int u = cv.u;
  return (u + 0x7fffu + ((u >> 16) & 1u)) >> 16;   // RNE f32 -> bf16
}
__device__ __forceinline__ unsigned int packbf(float a, float b) {
  return f2bf(a) | (f2bf(b) << 16);
}

// ---------------- weight prep: f32 OIHW -> bf16 [o][k = n*128 + c] ----------------
__global__ void prep_weights(const float* __restrict__ w_conv,
                             const float* __restrict__ w_off,
                             unsigned short* __restrict__ wmain,
                             unsigned short* __restrict__ woff) {
  int i = blockIdx.x * blockDim.x + threadIdx.x;
  if (i < kOutC * kK) {
    int o = i / kK, k = i - o * kK;
    int n = k >> 7, c = k & 127;
    wmain[i] = (unsigned short)f2bf(w_conv[(o * kInC + c) * kTaps + n]);
  } else {
    int i2 = i - kOutC * kK;
    if (i2 < 32 * kK) {                       // 18 real rows padded to 32
      int j = i2 / kK, k = i2 - j * kK;
      int n = k >> 7, c = k & 127;
      float v = 0.f;
      if (j < 18) {                           // fold cidx reorder into weights
        int js = (j < 9) ? 2 * j : 2 * (j - 9) + 1;
        v = w_off[(js * kInC + c) * kTaps + n];
      }
      woff[i2] = (unsigned short)f2bf(v);
    }
  }
}

// ---------------- offset conv (128 -> 18ch, pad 1) as implicit GEMM ----------------
__global__ __launch_bounds__(256)
void offset_conv(const float* __restrict__ x,
                 const unsigned int* __restrict__ woff,   // bf16 pairs, dword view
                 const float* __restrict__ b_off,
                 float* __restrict__ offw) {
  __shared__ __attribute__((aligned(16))) unsigned int Bt[2][kNTo][kStr];
  __shared__ unsigned int syx[kNTo];
  const int tid   = threadIdx.x;
  const int lane  = tid & 31;
  const int wave  = tid >> 5;
  const int mtile = wave & 1;               // 0..1 (M rows 0..31, 18 real)
  const int ntile = wave >> 1;              // 0..3 (N sub-tiles of 16)
  const int pid0  = blockIdx.x * kNTo;
  const int b     = pid0 / kHW;
  const int pixb  = pid0 - b * kHW;         // linear pixel base inside batch
  const float* xb = x + (size_t)b * kInC * kHW;

  if (tid < kNTo) {
    int pix = pixb + tid;
    int yy  = pix / kW;
    syx[tid] = (unsigned)yy | ((unsigned)(pix - yy * kW) << 16);
  }

  v8f acc = {};
  const int colN  = lane & 15;
  const int hi    = lane >> 4;
  const int dbase = hi * 4;
  const unsigned int* wr0 = woff + (size_t)(mtile * 16 + colN) * kKd;

  const int bN  = tid & (kNTo - 1);          // builder: pixel column
  const int bp0 = tid >> 6;                  // builder: base K-pair (0..3), units p = bp0+4u

  auto buildB = [&](int chunk, int buf) {
    const int n  = chunk >> 2;
    const int c0 = (chunk & 3) << 5;
    const int dy = n / 3 - 1, dx = n % 3 - 1;
    unsigned yx = syx[bN];
    int yy = (int)(yx & 0xffffu) + dy;
    int xx = (int)(yx >> 16) + dx;
    float v0[4] = {0.f, 0.f, 0.f, 0.f}, v1[4] = {0.f, 0.f, 0.f, 0.f};
    if (yy >= 0 && yy < kH && xx >= 0 && xx < kW) {
      const float* xc = xb + (size_t)(c0 + 2 * bp0) * kHW + yy * kW + xx;
      #pragma unroll
      for (int u = 0; u < 4; ++u) {           // one clause, immediate offsets
        v0[u] = xc[(size_t)u * 8 * kHW];
        v1[u] = xc[(size_t)u * 8 * kHW + kHW];
      }
    }
    #pragma unroll
    for (int u = 0; u < 4; ++u) Bt[buf][bN][bp0 + 4 * u] = packbf(v0[u], v1[u]);
  };

  __syncthreads();
  buildB(0, 0);
  __syncthreads();

  for (int chunk = 0; chunk < kChunks; ++chunk) {
    const int cur = chunk & 1;
    const int n = chunk >> 2, c0 = (chunk & 3) << 5;
    union { u32x4 q[2]; v16bf v; } afr, bfr;
    const unsigned int* wr = wr0 + ((n * 128 + c0) >> 1);
    afr.q[0] = *(const u32x4*)(wr + 2 * dbase);
    afr.q[1] = *(const u32x4*)(wr + 4 + 2 * dbase);
    const u32x4* rr = (const u32x4*)(&Bt[cur][ntile * 16 + colN][0]);
    bfr.q[0] = rr[hi * 2];
    bfr.q[1] = rr[hi * 2 + 1];
    if (chunk + 2 < kChunks) {
      int n2 = (chunk + 2) >> 2, c2 = ((chunk + 2) & 3) << 5;
      __builtin_prefetch(wr0 + ((n2 * 128 + c2) >> 1), 0, 0);
    }
    if (chunk + 1 < kChunks) buildB(chunk + 1, cur ^ 1);   // overlap with WMMA
    acc = __builtin_amdgcn_wmma_f32_16x16x32_bf16(false, afr.v, false, bfr.v,
                                                  (short)0, acc, false, false);
    __syncthreads();
  }

  #pragma unroll
  for (int r = 0; r < 8; ++r) {
    int j = mtile * 16 + r + hi * 8;          // C layout: M = r (+8 for hi lanes)
    if (j < 18) {
      int js = (j < 9) ? 2 * j : 2 * (j - 9) + 1;
      offw[((size_t)b * 18 + j) * kHW + pixb + ntile * 16 + colN] = acc[r] + b_off[js];
    }
  }
}

// ---------------- sampling metadata: plane offsets + pad-masked bilinear weights ----------------
__global__ void build_meta(const float* __restrict__ offw, Meta* __restrict__ meta) {
  int t = blockIdx.x * blockDim.x + threadIdx.x;
  if (t >= kBat * kTaps * kHW) return;
  int b   = t / (kTaps * kHW);
  int r0  = t - b * kTaps * kHW;
  int n   = r0 / kHW;
  int pix = r0 - n * kHW;
  int y   = pix / kW, x = pix - y * kW;
  float offy = offw[((size_t)b * 18 + n) * kHW + pix];
  float offx = offw[((size_t)b * 18 + n + 9) * kHW + pix];
  float pyr = (float)(y + 1) + (float)(n / 3 - 1) + offy;   // padded coords, ph=pw=98
  float pxr = (float)(x + 1) + (float)(n % 3 - 1) + offx;
  float fy = floorf(pyr), fx = floorf(pxr);
  float lty = fminf(fmaxf(fy,       0.f), 97.f);
  float ltx = fminf(fmaxf(fx,       0.f), 97.f);
  float rby = fminf(fmaxf(fy + 1.f, 0.f), 97.f);
  float rbx = fminf(fmaxf(fx + 1.f, 0.f), 97.f);
  float py = (pyr < 1.f || pyr > 96.f) ? fy : pyr;          // reference border masking
  float px = (pxr < 1.f || pxr > 96.f) ? fx : pxr;
  py = fminf(fmaxf(py, 0.f), 97.f);
  px = fminf(fmaxf(px, 0.f), 97.f);
  float glt = (1.f + (lty - py)) * (1.f + (ltx - px));
  float grb = (1.f - (rby - py)) * (1.f - (rbx - px));
  float glb = (1.f + (lty - py)) * (1.f - (rbx - px));
  float grt = (1.f - (rby - py)) * (1.f + (ltx - px));
  bool bly = (lty < 0.5f) || (lty > 96.5f);                 // corner on the zero-pad ring
  bool blx = (ltx < 0.5f) || (ltx > 96.5f);
  bool bry = (rby < 0.5f) || (rby > 96.5f);
  bool brx = (rbx < 0.5f) || (rbx > 96.5f);
  int iy0 = min(max((int)lty - 1, 0), 95);                  // padded -> unpadded coords
  int ix0 = min(max((int)ltx - 1, 0), 95);
  int iy1 = min(max((int)rby - 1, 0), 95);
  int ix1 = min(max((int)rbx - 1, 0), 95);
  Meta m;
  m.w0 = (bly || blx) ? 0.f : glt;                          // (lt_y, lt_x)
  m.w1 = (bry || brx) ? 0.f : grb;                          // (rb_y, rb_x)
  m.w2 = (bly || brx) ? 0.f : glb;                          // (lt_y, rb_x)
  m.w3 = (bry || blx) ? 0.f : grt;                          // (rb_y, lt_x)
  m.a00 = iy0 * kW + ix0;
  m.a11 = iy1 * kW + ix1;
  m.a01 = iy0 * kW + ix1;
  m.a10 = iy1 * kW + ix0;
  meta[t] = m;
}

// ---------------- main deformable conv as implicit GEMM (M=128, K=1152, N=pixels) ----------------
__global__ __launch_bounds__(256)
void deform_conv_main(const float* __restrict__ x,
                      const unsigned int* __restrict__ wmain,   // bf16 pairs, dword view
                      const Meta* __restrict__ meta,
                      float* __restrict__ out) {
  __shared__ __attribute__((aligned(16))) unsigned int Bt[2][kNT][kStr];
  __shared__ __attribute__((aligned(16))) Meta smeta[kTaps][kNT];
  const int tid  = threadIdx.x;
  const int lane = tid & 31;
  const int wave = tid >> 5;                 // M tile 0..7 (all 128 out channels)
  const int pid0 = blockIdx.x * kNT;
  const int b    = pid0 / kHW;
  const int pixb = pid0 - b * kHW;
  const float* xb = x + (size_t)b * kInC * kHW;

#if HAVE_ASYNC_LDS
  // async DMA: 9 rows x (32 Meta = 1 KB) -> contiguous 9 KB in LDS, b128 granules
  for (int i = tid; i < kTaps * (kNT * (int)sizeof(Meta) / 16); i += 256) {   // 576 granules
    int n = i >> 6, g16 = i & 63;
    const char* g = (const char*)(meta + ((size_t)b * kTaps + n) * kHW + pixb) + g16 * 16;
    char* l = (char*)(&smeta[0][0]) + i * 16;
    __builtin_amdgcn_global_load_async_to_lds_b128((i32x4v*)(void*)g, (as3_i32x4v*)l, 0, 0);
  }
  __builtin_amdgcn_s_wait_asynccnt(0);
#else
  for (int i = tid; i < kTaps * kNT; i += 256) {
    int n = i >> 5, j = i & (kNT - 1);
    smeta[n][j] = meta[((size_t)b * kTaps + n) * kHW + pixb + j];
  }
#endif

  v8f acc0 = {}, acc1 = {};
  const int colN  = lane & 15;
  const int hi    = lane >> 4;
  const int dbase = hi * 4;
  const unsigned int* wr0 = wmain + (size_t)(wave * 16 + colN) * kKd;

  const int bN = tid & (kNT - 1);            // builder: pixel column (0..31)
  const int bp = tid >> 5;                   // builder: K-pairs bp, bp+4 -> LDS dwords bp, bp+8

  auto buildB = [&](int chunk, int buf) {
    const int n  = chunk >> 2;
    const int c0 = (chunk & 3) << 5;
    Meta m = smeta[n][bN];
    const float* base = xb + (size_t)(c0 + 2 * bp) * kHW;
    const float* p00 = base + m.a00;          // 4 base addrs, 16 loads via imm offsets
    const float* p11 = base + m.a11;
    const float* p01 = base + m.a01;
    const float* p10 = base + m.a10;
    float v00 = m.w0 * p00[0]       + m.w1 * p11[0]       + m.w2 * p01[0]       + m.w3 * p10[0];
    float v01 = m.w0 * p00[kHW]     + m.w1 * p11[kHW]     + m.w2 * p01[kHW]     + m.w3 * p10[kHW];
    float v10 = m.w0 * p00[8 * kHW] + m.w1 * p11[8 * kHW] + m.w2 * p01[8 * kHW] + m.w3 * p10[8 * kHW];
    float v11 = m.w0 * p00[9 * kHW] + m.w1 * p11[9 * kHW] + m.w2 * p01[9 * kHW] + m.w3 * p10[9 * kHW];
    Bt[buf][bN][bp]     = packbf(v00, v01);
    Bt[buf][bN][bp + 8] = packbf(v10, v11);
  };

  __syncthreads();
  buildB(0, 0);
  __syncthreads();

  for (int chunk = 0; chunk < kChunks; ++chunk) {
    const int cur = chunk & 1;
    const int n = chunk >> 2, c0 = (chunk & 3) << 5;
    union { u32x4 q[2]; v16bf v; } afr, bfr0, bfr1;
    const unsigned int* wr = wr0 + ((n * 128 + c0) >> 1);
    afr.q[0] = *(const u32x4*)(wr + 2 * dbase);             // global_load_b128 x2
    afr.q[1] = *(const u32x4*)(wr + 4 + 2 * dbase);
    const u32x4* r0 = (const u32x4*)(&Bt[cur][colN][0]);    // ds_load_b128 x2 per frag
    const u32x4* r1 = (const u32x4*)(&Bt[cur][16 + colN][0]);
    bfr0.q[0] = r0[hi * 2]; bfr0.q[1] = r0[hi * 2 + 1];
    bfr1.q[0] = r1[hi * 2]; bfr1.q[1] = r1[hi * 2 + 1];
    if (chunk + 2 < kChunks) {                              // gfx1250 global_prefetch
      int n2 = (chunk + 2) >> 2, c2 = ((chunk + 2) & 3) << 5;
      __builtin_prefetch(wr0 + ((n2 * 128 + c2) >> 1), 0, 0);
    }
    if (chunk + 1 < kChunks) buildB(chunk + 1, cur ^ 1);    // overlap build with WMMA
    acc0 = __builtin_amdgcn_wmma_f32_16x16x32_bf16(false, afr.v, false, bfr0.v,
                                                   (short)0, acc0, false, false);
    acc1 = __builtin_amdgcn_wmma_f32_16x16x32_bf16(false, afr.v, false, bfr1.v,
                                                   (short)0, acc1, false, false);
    __syncthreads();
  }

  float* ob = out + (size_t)b * kOutC * kHW + pixb;
  #pragma unroll
  for (int r = 0; r < 8; ++r) {
    int o = wave * 16 + r + hi * 8;
    ob[(size_t)o * kHW + colN]      = acc0[r];
    ob[(size_t)o * kHW + 16 + colN] = acc1[r];
  }
}

// ---------------- launch ----------------
extern "C" void kernel_launch(void* const* d_in, const int* in_sizes, int n_in,
                              void* d_out, int out_size, void* d_ws, size_t ws_size,
                              hipStream_t stream) {
  const float* x      = (const float*)d_in[0];
  const float* w_off  = (const float*)d_in[1];
  const float* b_off  = (const float*)d_in[2];
  const float* w_conv = (const float*)d_in[3];
  float* out = (float*)d_out;

  char* ws = (char*)d_ws;
  unsigned short* wmain = (unsigned short*)ws;                         // 128*1152*2 = 294912 B
  unsigned short* woff  = (unsigned short*)(ws + 294912);              //  32*1152*2 =  73728 B
  float* offw           = (float*)(ws + 294912 + 73728);               // 4*18*9216*4 = 2654208 B
  Meta*  meta           = (Meta*)(ws + 294912 + 73728 + 2654208);      // 4*9*9216*32 = 10616832 B

  prep_weights<<<(kOutC * kK + 32 * kK + 255) / 256, 256, 0, stream>>>(w_conv, w_off, wmain, woff);
  offset_conv<<<kBat * kHW / kNTo, 256, 0, stream>>>(x, (const unsigned int*)woff, b_off, offw);
  build_meta<<<(kBat * kTaps * kHW + 255) / 256, 256, 0, stream>>>(offw, meta);
  deform_conv_main<<<kBat * kHW / kNT, 256, 0, stream>>>(x, (const unsigned int*)wmain, meta, out);
}
